// TemplateMatcher_62053687492842
// MI455X (gfx1250) — matrix operand order repaired and checked by reading, not compiled, not run on hardware
//
#include <hip/hip_runtime.h>
#include <hip/hip_bf16.h>
#include <math.h>

typedef __attribute__((ext_vector_type(16))) __bf16 v16bf;
typedef __attribute__((ext_vector_type(8)))  __bf16 bf16x8;
typedef __attribute__((ext_vector_type(8)))  float  v8f;

#define KPAD   272          // 257 clusters padded to 17*16
#define BIGV   1e30f

// workspace layout (float offsets)
#define WS_CNT   0           // [272] cluster counts
#define WS_SSQ   272         // [272] sum of squares (scalar per cluster)
#define WS_TN    544         // [272] ||template||^2
#define WS_THR   816         // [272] (1.5*std)^2 if valid else -1
#define WS_SUMS  1088        // [272*64] per-feature sums
#define WS_MEANS 18496       // bf16[272*64] template matrix starts here
#define WS_ZERO_FLOATS 18496

__device__ inline __bf16 f2bf(float x) { return (__bf16)x; }

// ---------------- K0: zero accumulators ----------------
__global__ void tm_zero(float* __restrict__ ws, int n) {
  int i = blockIdx.x * blockDim.x + threadIdx.x;
  int stride = gridDim.x * blockDim.x;
  for (; i < n; i += stride) ws[i] = 0.0f;
}

// ---------------- K1: segmented stats (wave per spike) ----------------
__global__ __launch_bounds__(256)
void tm_stats(const float* __restrict__ spikes, const int* __restrict__ sidx,
              float* __restrict__ ws, int M) {
  __shared__ float lsum[256 * 64];     // clusters 1..256, 64 features, 64KB
  float* gcnt = ws + WS_CNT;
  float* gssq = ws + WS_SSQ;
  float* gsum = ws + WS_SUMS;

  for (int e = threadIdx.x; e < 256 * 64; e += blockDim.x) lsum[e] = 0.0f;
  __syncthreads();

  int lane = threadIdx.x & 31;
  int wid  = (blockIdx.x * blockDim.x + threadIdx.x) >> 5;
  int nw   = (gridDim.x * blockDim.x) >> 5;

  for (int i = wid; i < M; i += nw) {
    int c = sidx[i];                               // wave-uniform
    float a = spikes[(size_t)i * 64 + lane];
    float b = spikes[(size_t)i * 64 + 32 + lane];
    float s = a * a + b * b;
    #pragma unroll
    for (int off = 16; off >= 1; off >>= 1) s += __shfl_xor(s, off);
    if (c != 0) {                                  // cluster 0 never used downstream
      atomicAdd(&lsum[(c - 1) * 64 + lane], a);
      atomicAdd(&lsum[(c - 1) * 64 + 32 + lane], b);
      if (lane == 0) {
        atomicAdd(&gssq[c], s);
        atomicAdd(&gcnt[c], 1.0f);
      }
    }
  }
  __syncthreads();
  for (int e = threadIdx.x; e < 256 * 64; e += blockDim.x) {
    float v = lsum[e];
    if (v != 0.0f) atomicAdd(&gsum[64 + e], v);    // cluster c row at gsum[c*64]
  }
}

// ---------------- K2: finalize templates, median filter, squared thresholds ----------------
__global__ __launch_bounds__(256)
void tm_finalize(float* __restrict__ ws) {
  const float* gcnt = ws + WS_CNT;
  const float* gssq = ws + WS_SSQ;
  const float* gsum = ws + WS_SUMS;
  float* tnA  = ws + WS_TN;
  float* thrA = ws + WS_THR;
  __bf16* mB  = (__bf16*)(ws + WS_MEANS);

  __shared__ float s_std[KPAD];
  __shared__ unsigned char s_val[KPAD];
  __shared__ float svals[512];
  __shared__ int scnt;
  __shared__ float s_med;
  if (threadIdx.x == 0) scnt = 0;

  for (int c = threadIdx.x; c < KPAD; c += blockDim.x) {
    float tn = 0.0f, stdv = 0.0f;
    bool valid = false;
    if (c >= 1 && c <= 256) {
      float cnt  = gcnt[c];
      float safe = fmaxf(cnt, 1.0f);
      #pragma unroll 4
      for (int f = 0; f < 64; ++f) {
        float m = gsum[c * 64 + f] / safe;
        tn += m * m;
        mB[c * 64 + f] = f2bf(m);
      }
      stdv  = sqrtf(fmaxf(gssq[c] / safe - tn, 0.0f));  // sum over features of var
      valid = (cnt > 0.0f);
    } else {
      for (int f = 0; f < 64; ++f) mB[c * 64 + f] = f2bf(0.0f);
    }
    tnA[c]   = tn;
    s_std[c] = stdv;
    s_val[c] = valid ? 1 : 0;
  }
  __syncthreads();

  for (int c = threadIdx.x; c < KPAD; c += blockDim.x)
    if (s_val[c]) { int p = atomicAdd(&scnt, 1); svals[p] = s_std[c]; }
  __syncthreads();
  int V = scnt;
  for (int p = threadIdx.x; p < 512; p += blockDim.x)
    if (p >= V) svals[p] = BIGV;
  __syncthreads();

  // odd-even transposition sort of 512 elements (<=256 real)
  for (int ph = 0; ph < 512; ++ph) {
    int i = 2 * (int)threadIdx.x + (ph & 1);
    if (i + 1 < 512) {
      float x = svals[i], y = svals[i + 1];
      if (y < x) { svals[i] = y; svals[i + 1] = x; }
    }
    __syncthreads();
  }
  if (threadIdx.x == 0) {
    s_med = (V == 0) ? 0.0f
          : ((V & 1) ? svals[V / 2] : 0.5f * (svals[V / 2 - 1] + svals[V / 2]));
  }
  __syncthreads();
  float med = s_med;
  for (int c = threadIdx.x; c < KPAD; c += blockDim.x) {
    bool v = s_val[c] && (s_std[c] <= 3.0f * med);
    float t = 1.5f * s_std[c];
    thrA[c] = v ? t * t : -1.0f;       // squared threshold; -1 => always masked
  }
}

// ---------------- K3: WMMA matmul + masked argmin in d^2 domain ----------------
__global__ __launch_bounds__(256)
void tm_match(const float* __restrict__ spikes, const int* __restrict__ sidx,
              const unsigned char* __restrict__ midx,
              const float* __restrict__ ws,
              float* __restrict__ out, int M, int ntiles) {
  const float*  tnA  = ws + WS_TN;
  const float*  thrA = ws + WS_THR;
  const __bf16* mB   = (const __bf16*)(ws + WS_MEANS);

  int lane = threadIdx.x & 31;
  int wid  = (blockIdx.x * blockDim.x + threadIdx.x) >> 5;
  if (wid >= ntiles) return;                        // wave-uniform: EXEC all-ones below
  int m0 = wid * 16;

  bool lo = lane < 16;
  int  r  = lane & 15;
  int  row = m0 + r; if (row >= M) row = M - 1;
  const float* rowp = spikes + (size_t)row * 64;
  int kofs = lo ? 0 : 8;

  // A fragments (16x32 bf16 each): lane<16 K{0..7,16..23}+b*32 ; lane>=16 K{8..15,24..31}+b*32
  float sn = 0.0f;
  v16bf a0 = {}, a1 = {};
  #pragma unroll
  for (int b = 0; b < 2; ++b) {
    int base1 = b * 32 + kofs;
    int base2 = b * 32 + 16 + kofs;
    float4 f0 = *(const float4*)(rowp + base1);
    float4 f1 = *(const float4*)(rowp + base1 + 4);
    float4 f2 = *(const float4*)(rowp + base2);
    float4 f3 = *(const float4*)(rowp + base2 + 4);
    sn += f0.x*f0.x + f0.y*f0.y + f0.z*f0.z + f0.w*f0.w;
    sn += f1.x*f1.x + f1.y*f1.y + f1.z*f1.z + f1.w*f1.w;
    sn += f2.x*f2.x + f2.y*f2.y + f2.z*f2.z + f2.w*f2.w;
    sn += f3.x*f3.x + f3.y*f3.y + f3.z*f3.z + f3.w*f3.w;
    v16bf& aa = b ? a1 : a0;
    aa[0]=f2bf(f0.x); aa[1]=f2bf(f0.y); aa[2]=f2bf(f0.z); aa[3]=f2bf(f0.w);
    aa[4]=f2bf(f1.x); aa[5]=f2bf(f1.y); aa[6]=f2bf(f1.z); aa[7]=f2bf(f1.w);
    aa[8]=f2bf(f2.x); aa[9]=f2bf(f2.y); aa[10]=f2bf(f2.z); aa[11]=f2bf(f2.w);
    aa[12]=f2bf(f3.x); aa[13]=f2bf(f3.y); aa[14]=f2bf(f3.z); aa[15]=f2bf(f3.w);
  }
  sn += __shfl_xor(sn, 16);                         // lane l holds ||spike[m0+(l&15)]||^2

  // hoist row-norm broadcasts OUT of the chunk loop (loop-invariant DS ops)
  float snj[8];
  #pragma unroll
  for (int j = 0; j < 8; ++j) snj[j] = __shfl(sn, lo ? j : (j + 8));

  float best[8]; int bidx[8];
  #pragma unroll
  for (int j = 0; j < 8; ++j) { best[j] = BIGV; bidx[j] = 0; }

  int koffB = lo ? 0 : 8;
  for (int ch = 0; ch < KPAD / 16; ++ch) {
    int n = ch * 16 + (lane & 15);                  // this lane's cluster column
    const __bf16* bp = mB + (size_t)n * 64;
    bf16x8 p00 = *(const bf16x8*)(bp + koffB);
    bf16x8 p01 = *(const bf16x8*)(bp + koffB + 16);
    bf16x8 p10 = *(const bf16x8*)(bp + 32 + koffB);
    bf16x8 p11 = *(const bf16x8*)(bp + 32 + koffB + 16);
    v16bf b0, b1;
    #pragma unroll
    for (int i = 0; i < 8; ++i) {
      b0[i] = p00[i]; b0[i + 8] = p01[i];
      b1[i] = p10[i]; b1[i + 8] = p11[i];
    }
    v8f c = {};
    c = __builtin_amdgcn_wmma_f32_16x16x32_bf16(false, a0, false, b0, (short)0, c, false, false);
    c = __builtin_amdgcn_wmma_f32_16x16x32_bf16(false, a1, false, b1, (short)0, c, false, false);

    float tn   = tnA[n];
    float thr2 = thrA[n];
    #pragma unroll
    for (int j = 0; j < 8; ++j) {                   // VGPR j -> row m0+j (+8 for hi lanes)
      float d2 = fmaf(-2.0f, c[j], snj[j] + tn);
      d2 = fmaxf(d2, 0.0f);
      d2 = (d2 > thr2) ? BIGV : d2;                 // thr2=-1 => masked (d2>=0 > -1)
      if (d2 < best[j]) { best[j] = d2; bidx[j] = n; }
    }
  }

  // reduce across the 16 column-lanes of each half; first-index tie-break
  #pragma unroll
  for (int j = 0; j < 8; ++j) {
    #pragma unroll
    for (int off = 8; off >= 1; off >>= 1) {
      float ov = __shfl_xor(best[j], off);
      int   oi = __shfl_xor(bidx[j], off);
      if (ov < best[j] || (ov == best[j] && oi < bidx[j])) { best[j] = ov; bidx[j] = oi; }
    }
  }

  float* outSort  = out;
  float* outMatch = out + (size_t)M;
  float* outMin   = out + 2 * (size_t)M;
  if (lane == 0 || lane == 16) {
    int rbase = m0 + (lo ? 0 : 8);
    for (int j = 0; j < 8; ++j) {
      int rr = rbase + j;
      if (rr < M) {
        float b2 = best[j];
        float mv = (b2 >= 1e29f) ? BIGV : sqrtf(b2);          // sentinel stays exactly 1e30
        int   mi = (b2 >= 65536.0f) ? 0 : bidx[j];            // dist>=256 <=> d2>=256^2
        int   so = sidx[rr];
        bool  unm = (so == 0);
        int   ns  = unm ? mi : so;
        float nm  = unm ? ((mi != 0) ? 1.0f : 0.0f) : (midx[rr] ? 1.0f : 0.0f);
        outSort[rr]  = (float)ns;
        outMatch[rr] = nm;
        outMin[rr]   = mv;
      }
    }
  }
}

extern "C" void kernel_launch(void* const* d_in, const int* in_sizes, int n_in,
                              void* d_out, int out_size, void* d_ws, size_t ws_size,
                              hipStream_t stream) {
  const float*         spikes = (const float*)d_in[0];
  const int*           sidx   = (const int*)d_in[1];
  const unsigned char* midx   = (const unsigned char*)d_in[2];
  float* ws  = (float*)d_ws;
  float* out = (float*)d_out;
  int M = in_sizes[1];                  // number of spikes

  hipLaunchKernelGGL(tm_zero, dim3(32), dim3(256), 0, stream, ws, (int)WS_ZERO_FLOATS);
  hipLaunchKernelGGL(tm_stats, dim3(256), dim3(256), 0, stream, spikes, sidx, ws, M);
  hipLaunchKernelGGL(tm_finalize, dim3(1), dim3(256), 0, stream, ws);

  int ntiles  = (M + 15) / 16;
  int nblocks = (ntiles + 7) / 8;       // 8 waves per 256-thread block
  hipLaunchKernelGGL(tm_match, dim3(nblocks), dim3(256), 0, stream,
                     spikes, sidx, midx, ws, out, M, ntiles);
}
